// BahdanauAttention_4982162063532
// MI455X (gfx1250) — matrix operand order repaired and checked by reading, not compiled, not run on hardware
//
#include <hip/hip_runtime.h>
#include <hip/hip_bf16.h>

// ---------------------------------------------------------------------------
// Bahdanau additive attention for MI455X (gfx1250, wave32).
//   q = query @ W1                  (WMMA bf16, f32 accum)
//   k = value @ W2                  (WMMA bf16, f32 accum)
//   scores = sum_u scale*tanh(q+k)  (f32 VALU, native tanh, 8-row k reuse,
//                                    q-tile staged via async LDS loads)
//   attn = softmax(mask(scores))    (wave32 shfl reductions)
//   context = attn @ value          (batched WMMA bf16, f32 accum)
// Output: [context (16*128*512 f32) | attn (16*128*256 f32)]
// ---------------------------------------------------------------------------

typedef __attribute__((ext_vector_type(16))) __bf16 v16bf;
typedef __attribute__((ext_vector_type(8)))  __bf16 v8bf;
typedef __attribute__((ext_vector_type(8)))  float  v8f;

#define BN 16
#define TT 128
#define SS 256
#define DD 512
#define UU 512
#define TROW 8   // t-rows per scores block == waves per block (256/32)

__device__ __forceinline__ float fast_tanh(float x) {
#if __has_builtin(__builtin_amdgcn_tanhf)
  return __builtin_amdgcn_tanhf(x);
#elif __has_builtin(__builtin_amdgcn_tanh_f32)
  return __builtin_amdgcn_tanh_f32(x);
#else
  return tanhf(x);
#endif
}

__device__ __forceinline__ void async_wait0() {
#if __has_builtin(__builtin_amdgcn_s_wait_asynccnt)
  __builtin_amdgcn_s_wait_asynccnt(0);
#else
  asm volatile("s_wait_asynccnt 0x0" ::: "memory");
#endif
}

// Async global->LDS copy of 16B per lane (CDNA5 ASYNCcnt path).
__device__ __forceinline__ void async_g2l_b128(void* lds_ptr, const void* gptr) {
  unsigned int       l = (unsigned int)(unsigned long long)(uintptr_t)lds_ptr;
  unsigned long long g = (unsigned long long)(uintptr_t)gptr;
  asm volatile("global_load_async_to_lds_b128 %0, %1, off"
               :: "v"(l), "v"(g) : "memory");
}

// ---- f32 -> bf16 row-major convert -----------------------------------------
__global__ void convert_bf16_kernel(const float* __restrict__ in,
                                    __bf16* __restrict__ out, int n) {
  int i = blockIdx.x * blockDim.x + threadIdx.x;
  if (i < n) out[i] = (__bf16)in[i];
}

// ---- f32 -> bf16 transpose-convert: out[b][c][r] = in[b][r][c] -------------
__global__ void transpose_bf16_kernel(const float* __restrict__ in,
                                      __bf16* __restrict__ out,
                                      int B, int R, int C) {
  int i = blockIdx.x * blockDim.x + threadIdx.x;
  int total = B * R * C;
  if (i >= total) return;
  int b = i / (R * C);
  int rem = i - b * R * C;
  int r = rem / C;
  int c = rem - r * C;
  out[(long)b * R * C + (long)c * R + r] = (__bf16)in[i];
}

// ---- generic batched GEMM: C[M,N] = A[M,K](row) * B[N,K](col-major) --------
// One wave computes one 16x16 C tile; K-loop in steps of 32 via
// v_wmma_f32_16x16x32_bf16. A and B fragments both use contiguous k order, k = 16*half + s.
__global__ void wmma_gemm_bf16_kernel(const __bf16* __restrict__ A,
                                      const __bf16* __restrict__ B,
                                      float* __restrict__ C,
                                      int M, int N, int K, int batch,
                                      long sA, long sB, long sC) {
  int wid  = (blockIdx.x * blockDim.x + threadIdx.x) >> 5;
  int lane = threadIdx.x & 31;
  int tilesM = M >> 4, tilesN = N >> 4;
  int tilesPerBatch = tilesM * tilesN;
  if (wid >= batch * tilesPerBatch) return;   // wave-uniform: EXEC stays all-1

  int b   = wid / tilesPerBatch;
  int rem = wid - b * tilesPerBatch;
  int tm  = rem / tilesN;
  int tn  = rem - tm * tilesN;

  const __bf16* Ab = A + (long)b * sA;
  const __bf16* Bb = B + (long)b * sB;
  float*        Cb = C + (long)b * sC;

  int half = lane >> 4;        // 0: lanes 0-15, 1: lanes 16-31
  int r    = lane & 15;
  long arow = (long)((tm << 4) + r) * K;   // A row for this lane
  long bcol = (long)((tn << 4) + r) * K;   // B column for this lane

  v8f acc = {};
  for (int k0 = 0; k0 < K; k0 += 32) {
    const __bf16* pa = Ab + arow + k0 + half * 16;
    v8bf alo = *(const v8bf*)(pa);
    v8bf ahi = *(const v8bf*)(pa + 8);
    v16bf afrag = __builtin_shufflevector(alo, ahi,
        0, 1, 2, 3, 4, 5, 6, 7, 8, 9, 10, 11, 12, 13, 14, 15);
    v16bf bfrag = *(const v16bf*)(Bb + bcol + k0 + half * 16);
    __builtin_prefetch(pa + 32, 0, 0);   // global_prefetch_b8 next K-slab
    acc = __builtin_amdgcn_wmma_f32_16x16x32_bf16(
        false, afrag, false, bfrag, (short)0, acc, false, false);
  }

  int row0 = (tm << 4) + half * 8;
  float* cp = Cb + (long)row0 * N + (tn << 4) + r;
#pragma unroll
  for (int j = 0; j < 8; ++j) cp[(long)j * N] = acc[j];
}

// ---- fused scores + mask + softmax -----------------------------------------
// block = (b, tile of TROW t-rows): 256 threads (8 waves), thread s in [0,256).
// Each float4 of k feeds 8 rows (32 tanh+FMA) -> 8x less k traffic, 8-way ILP
// into the TRANS pipe. q tile + scale staged in LDS via async loads.
__global__ void scores_softmax_kernel(const float* __restrict__ q,
                                      const float* __restrict__ k,
                                      const unsigned char* __restrict__ mask,
                                      const float* __restrict__ scale,
                                      float* __restrict__ attn_f32,
                                      __bf16* __restrict__ attn_bf) {
  int b  = blockIdx.x / (TT / TROW);
  int tt = blockIdx.x - b * (TT / TROW);
  int t0 = tt * TROW;
  int s  = threadIdx.x;            // 0..255

  __shared__ float qs[TROW * UU];  // 16 KB: 8 contiguous q rows
  __shared__ float sc[UU];         //  2 KB
  __shared__ float srow[TROW * SS];//  8 KB: scores for in-block softmax
  __shared__ float rowmax[TROW];
  __shared__ float rowsum[TROW];

  // --- stage q tile (8*512 contiguous floats) + scale via async LDS loads ---
  const float* qbase = q + ((long)(b * TT + t0)) * UU;
#pragma unroll
  for (int p = 0; p < (TROW * UU) / (SS * 4); ++p) {  // 4 passes of 16B/lane
    int idx = p * SS * 4 + s * 4;
    async_g2l_b128(&qs[idx], qbase + idx);
  }
  if (s < UU / 4) async_g2l_b128(&sc[s * 4], scale + s * 4);
  async_wait0();
  __syncthreads();

  // --- additive scores: acc[j] = sum_u sc[u]*tanh(qs[j][u] + k[s][u]) ---
  float acc[TROW];
#pragma unroll
  for (int j = 0; j < TROW; ++j) acc[j] = 0.0f;

  const float4* krow = (const float4*)(k + ((long)(b * SS + s)) * UU);
  for (int u4 = 0; u4 < UU / 4; ++u4) {
    float4 kv = krow[u4];
    int u = u4 * 4;
    float s0 = sc[u], s1 = sc[u + 1], s2 = sc[u + 2], s3 = sc[u + 3];
#pragma unroll
    for (int j = 0; j < TROW; ++j) {
      const float* qj = qs + j * UU + u;
      acc[j] = fmaf(s0, fast_tanh(qj[0] + kv.x), acc[j]);
      acc[j] = fmaf(s1, fast_tanh(qj[1] + kv.y), acc[j]);
      acc[j] = fmaf(s2, fast_tanh(qj[2] + kv.z), acc[j]);
      acc[j] = fmaf(s3, fast_tanh(qj[3] + kv.w), acc[j]);
    }
  }

  // --- mask, publish scores ---
  bool mk = mask[b * SS + s] != 0;
  float score[TROW];
#pragma unroll
  for (int j = 0; j < TROW; ++j) {
    score[j] = mk ? acc[j] : -1e9f;
    srow[j * SS + s] = score[j];
  }
  __syncthreads();

  // --- softmax stats: wave w reduces row w (wave32 shfl_xor) ---
  int w = s >> 5, lane = s & 31;
  {
    const float* row = srow + w * SS;
    float mx = -3.4e38f;
    for (int i = lane; i < SS; i += 32) mx = fmaxf(mx, row[i]);
#pragma unroll
    for (int off = 16; off > 0; off >>= 1)
      mx = fmaxf(mx, __shfl_xor(mx, off, 32));
    float sm = 0.0f;
    for (int i = lane; i < SS; i += 32) sm += __expf(row[i] - mx);
#pragma unroll
    for (int off = 16; off > 0; off >>= 1)
      sm += __shfl_xor(sm, off, 32);
    if (lane == 0) { rowmax[w] = mx; rowsum[w] = sm; }
  }
  __syncthreads();

  // --- normalize + emit f32 (output) and bf16 (for context WMMA) ---
#pragma unroll
  for (int j = 0; j < TROW; ++j) {
    float a = __expf(score[j] - rowmax[j]) * (1.0f / rowsum[j]);
    long idx = ((long)(b * TT + t0 + j)) * SS + s;
    attn_f32[idx] = a;
    attn_bf[idx]  = (__bf16)a;
  }
}

// ---------------------------------------------------------------------------
extern "C" void kernel_launch(void* const* d_in, const int* in_sizes, int n_in,
                              void* d_out, int out_size, void* d_ws, size_t ws_size,
                              hipStream_t stream) {
  const float*         query = (const float*)d_in[0];          // (16,128,512)
  const float*         value = (const float*)d_in[1];          // (16,256,512)
  const unsigned char* mask  = (const unsigned char*)d_in[2];  // (16,256) bool
  const float*         W1    = (const float*)d_in[3];          // (512,512)
  const float*         W2    = (const float*)d_in[4];          // (512,512)
  const float*         scale = (const float*)d_in[5];          // (512,)

  float* context  = (float*)d_out;                       // 16*128*512
  float* attn_f32 = (float*)d_out + (long)BN * TT * DD;  // 16*128*256

  // ---- workspace layout (bytes, 256-aligned) ----
  char* ws = (char*)d_ws;
  size_t off = 0;
  auto take = [&](size_t bytes) {
    char* p = ws + off;
    off = (off + bytes + 255) & ~(size_t)255;
    return p;
  };
  float*  qf32   = (float*)take((size_t)BN * TT * UU * 4);   // 4 MB
  float*  kf32   = (float*)take((size_t)BN * SS * UU * 4);   // 8 MB
  __bf16* qbf    = (__bf16*)take((size_t)BN * TT * DD * 2);  // 2 MB
  __bf16* vbf    = (__bf16*)take((size_t)BN * SS * DD * 2);  // 4 MB
  __bf16* w1t    = (__bf16*)take((size_t)DD * UU * 2);       // 0.5 MB (col-major)
  __bf16* w2t    = (__bf16*)take((size_t)DD * UU * 2);       // 0.5 MB (col-major)
  __bf16* vtbf   = (__bf16*)take((size_t)BN * DD * SS * 2);  // 4 MB (per-b value^T)
  __bf16* attnbf = (__bf16*)take((size_t)BN * TT * SS * 2);  // 1 MB

  // ---- stage 0: precision conversion + transposes ----
  {
    int n = BN * TT * DD;
    convert_bf16_kernel<<<(n + 255) / 256, 256, 0, stream>>>(query, qbf, n);
  }
  {
    int n = BN * SS * DD;
    convert_bf16_kernel<<<(n + 255) / 256, 256, 0, stream>>>(value, vbf, n);
  }
  {
    int n = DD * UU;
    transpose_bf16_kernel<<<(n + 255) / 256, 256, 0, stream>>>(W1, w1t, 1, DD, UU);
    transpose_bf16_kernel<<<(n + 255) / 256, 256, 0, stream>>>(W2, w2t, 1, DD, UU);
  }
  {
    int n = BN * SS * DD;
    transpose_bf16_kernel<<<(n + 255) / 256, 256, 0, stream>>>(value, vtbf, BN, SS, DD);
  }

  // ---- stage 1: projections (WMMA bf16) ----
  {  // q = query @ W1 : M=2048, N=512, K=512
    int waves = (BN * TT / 16) * (UU / 16);
    wmma_gemm_bf16_kernel<<<(waves + 7) / 8, 256, 0, stream>>>(
        qbf, w1t, qf32, BN * TT, UU, DD, 1, 0, 0, 0);
  }
  {  // k = value @ W2 : M=4096, N=512, K=512
    int waves = (BN * SS / 16) * (UU / 16);
    wmma_gemm_bf16_kernel<<<(waves + 7) / 8, 256, 0, stream>>>(
        vbf, w2t, kf32, BN * SS, UU, DD, 1, 0, 0, 0);
  }

  // ---- stage 2: additive scores + mask + softmax (dominant tanh work) ----
  scores_softmax_kernel<<<BN * (TT / TROW), SS, 0, stream>>>(
      qf32, kf32, mask, scale, attn_f32, attnbf);

  // ---- stage 3: context = attn @ value (batched WMMA bf16) ----
  {  // per b: M=128, N=512, K=256
    int waves = BN * (TT / 16) * (DD / 16);
    wmma_gemm_bf16_kernel<<<(waves + 7) / 8, 256, 0, stream>>>(
        attnbf, vtbf, context, TT, DD, SS, BN,
        (long)TT * SS, (long)DD * SS, (long)TT * DD);
  }
}
